// DeformableFeatureAlignment_17746804867794
// MI455X (gfx1250) — compile-verified, matrix operand order checked by reading, not run on hardware
//
#include <hip/hip_runtime.h>

// ---------------------------------------------------------------------------
// DeformableFeatureAlignment for MI455X (gfx1250, wave32, WMMA, async-LDS)
//   B=4, H=W=64, Cc=Cf=F=256, GROUPS=8, K=3x3
// All heavy stages are bf16 WMMA GEMMs (f32 accumulate):
//   - weights pre-converted once to bf16 and transposed to [N][K] so GEMM
//     B^T tiles are contiguous and streamed with global_load_async_to_lds_b128
//     (ASYNCcnt), overlapping with fused A-tile construction.
//   gemm<1,0>: fine_cal = (fine*(attn+1)) @ w_sel
//   gemm<2,1>: align(bf16) = concat(fine_cal, 2*coarse_up) @ w_off
//   gemm<5,2>: om = im2col3x3(align) @ w_om + b_om
//   gemm<4,3>: out = relu(deform_sample(coarse_up, om) @ w_dcn + b_dcn) + fine_cal
// ---------------------------------------------------------------------------

typedef __attribute__((ext_vector_type(16))) __bf16 v16bf;
typedef __attribute__((ext_vector_type(8)))  float  v8f;
typedef __attribute__((address_space(3))) unsigned short lds_ushort;

union Frag { v16bf v; uint4 q[2]; };

static __device__ __forceinline__ unsigned short f2bfbits(float f) {
    union { float f; unsigned u; } v; v.f = f;
    // round-to-nearest-even bf16
    unsigned r = (v.u + 0x7FFFu + ((v.u >> 16) & 1u)) >> 16;
    return (unsigned short)r;
}

// ---------------------------------------------------------------------------
// Weight prep: WT[NP][K] (bf16) = transpose(W[K][N]) with zero row padding.
// grid (NP, K/256), block 256. One-time cost (~2.1M elements total).
// ---------------------------------------------------------------------------
__global__ __launch_bounds__(256) void wprep_kernel(const float* __restrict__ W,
                                                    unsigned short* __restrict__ WT,
                                                    int K, int N) {
    int n = blockIdx.x;
    int k = blockIdx.y * 256 + threadIdx.x;
    unsigned short o = 0;
    if (n < N) o = f2bfbits(W[(size_t)k * N + n]);
    WT[(size_t)n * K + k] = o;
}

// ---------------------------------------------------------------------------
// Global mean pool over H,W:  pooled[b][c] = mean(fine[b,:,:,c])
// ---------------------------------------------------------------------------
__global__ __launch_bounds__(256) void pool_kernel(const float* __restrict__ fine,
                                                   float* __restrict__ pooled) {
    int b = blockIdx.x, cg = blockIdx.y;
    int t = threadIdx.x;
    int cl = t & 63;
    int sl = t >> 6;
    const float* base = fine + (size_t)b * 4096 * 256;
    float s = 0.f;
    for (int p = sl; p < 4096; p += 4) s += base[(size_t)p * 256 + cg * 64 + cl];
    __shared__ float red[256];
    red[t] = s;
    __syncthreads();
    if (t < 64)
        pooled[b * 256 + cg * 64 + t] =
            (red[t] + red[t + 64] + red[t + 128] + red[t + 192]) * (1.f / 4096.f);
}

// ---------------------------------------------------------------------------
// attn[b][n] = sigmoid( sum_k pooled[b][k] * w_att[k][n] )
// ---------------------------------------------------------------------------
__global__ __launch_bounds__(256) void attn_kernel(const float* __restrict__ pooled,
                                                   const float* __restrict__ w_att,
                                                   float* __restrict__ attn) {
    int b = blockIdx.x, n = threadIdx.x;
    __shared__ float ps[256];
    ps[n] = pooled[b * 256 + n];
    __syncthreads();
    float s = 0.f;
    #pragma unroll 4
    for (int k = 0; k < 256; ++k) s += ps[k] * w_att[k * 256 + n];
    attn[b * 256 + n] = 1.f / (1.f + __expf(-s));
}

// ---------------------------------------------------------------------------
// Bilinear 32x32 -> 64x64 upsample (half-pixel centers, edge clamp)
// ---------------------------------------------------------------------------
__global__ __launch_bounds__(256) void upsample_kernel(const float* __restrict__ coarse,
                                                       float* __restrict__ up) {
    int pix = blockIdx.x, b = blockIdx.y, c = threadIdx.x;
    int y = pix >> 6, x = pix & 63;
    float fy = fminf(fmaxf((y + 0.5f) * 0.5f - 0.5f, 0.f), 31.f);
    float fx = fminf(fmaxf((x + 0.5f) * 0.5f - 0.5f, 0.f), 31.f);
    int y0 = (int)fy, x0 = (int)fx;
    int y1 = min(y0 + 1, 31), x1 = min(x0 + 1, 31);
    float wy = fy - (float)y0, wx = fx - (float)x0;
    const float* base = coarse + (size_t)b * 1024 * 256;
    float v00 = base[((size_t)y0 * 32 + x0) * 256 + c];
    float v01 = base[((size_t)y0 * 32 + x1) * 256 + c];
    float v10 = base[((size_t)y1 * 32 + x0) * 256 + c];
    float v11 = base[((size_t)y1 * 32 + x1) * 256 + c];
    float top = v00 * (1.f - wx) + v01 * wx;
    float bot = v10 * (1.f - wx) + v11 * wx;
    up[((size_t)b * 4096 + pix) * 256 + c] = top * (1.f - wy) + bot * wy;
}

// ---------------------------------------------------------------------------
// Generic bf16 WMMA GEMM: C[M,N] = A[M,K] @ B[K,N]
// B supplied pre-transposed bf16: BT[NP][K], NP = N padded up to 128 multiple.
// Macro tile 128x128, 8 waves (2 M-halves x 4 N-strips), wave tile 64x32,
// K-step 32. B^T tile streamed via global_load_async_to_lds_b128 (ASYNCcnt),
// overlapped with fused A-tile construction.
//
// A_MODE: 0 plain f32    1 fine*(attn+1)    2 concat(fine_cal, 2*coarse_up)
//         3 bf16 direct  4 fused deformable sampling (om offsets + coarse_up)
//         5 fused 3x3 im2col over bf16 'align'
// E_MODE: 0 f32 store    1 bf16 store       2 +bias f32 store (N-guarded)
//         3 relu(+bias) + resid store f32
// ---------------------------------------------------------------------------
template<int A_MODE, int E_MODE>
__global__ __launch_bounds__(256) void gemm_wmma(
    const float* __restrict__ Af, const unsigned short* __restrict__ Ab,
    const float* __restrict__ attn, const float* __restrict__ fineCal,
    const float* __restrict__ coarseUp,
    const unsigned short* __restrict__ BT, const float* __restrict__ bias,
    const float* __restrict__ resid,
    float* __restrict__ outF, unsigned short* __restrict__ outB,
    int M, int K, int N)
{
    __shared__ __align__(16) unsigned short As[128][32];  // [m][k] bf16
    __shared__ __align__(16) unsigned short Bs[128][32];  // [n][k] bf16 (B^T tile)

    const int tid = threadIdx.x;
    const int m0 = blockIdx.x * 128;
    const int n0 = blockIdx.y * 128;
    const int wid = tid >> 5;
    const int lane = tid & 31;
    const int lj  = lane & 15;   // N (or M) index within 16
    const int lhi = lane >> 4;   // K-half select per ISA 16-bit layout
    const int wm = wid & 1;      // 64-row half
    const int wn = wid >> 1;     // 32-col strip

    v8f acc[4][2];
    {
        v8f z = 0.f;
        #pragma unroll
        for (int mi = 0; mi < 4; ++mi)
            #pragma unroll
            for (int ni = 0; ni < 2; ++ni) acc[mi][ni] = z;
    }

    const int atc = tid & 31;    // k within chunk (A load)
    const int atr = tid >> 5;    // row base (A load)
    // B^T async copy mapping: 128 rows x 64B tile; 16B per lane; 2 issues.
    const int brow = tid >> 2;   // row within 64-row half
    const int bseg = tid & 3;    // 16B segment within row

    for (int kb = 0; kb < K; kb += 32) {
        if (kb) __syncthreads();

        // ---- B^T tile: async DMA global(bf16) -> LDS, contiguous rows ----
        #pragma unroll
        for (int half = 0; half < 2; ++half) {
            int row = half * 64 + brow;
            const unsigned short* g =
                BT + (size_t)(n0 + row) * K + kb + bseg * 8;
            unsigned lds_addr =
                (unsigned)(unsigned long)(lds_ushort*)&Bs[row][bseg * 8];
            asm volatile("global_load_async_to_lds_b128 %0, %1, off"
                         :: "v"(lds_addr), "v"(g) : "memory");
        }

        // -------------------- A tile: 128 rows x 32 k --------------------
        if (A_MODE == 4) {
            // deformable bilinear sample of coarse_up; offsets/mask from om(=Af)
            // flat k = tap*256 + g*32 + c  ->  chunk = tap*8 + g
            const int chunk = kb >> 5;
            const int tap = chunk >> 3;
            const int g   = chunk & 7;
            const int ch  = g * 32 + atc;
            const int kyo = tap / 3 - 1, kxo = tap % 3 - 1;
            #pragma unroll 1
            for (int rr = 0; rr < 16; ++rr) {
                int r = atr + rr * 8, m = m0 + r;
                int b = m >> 12, pix = m & 4095;
                int y = pix >> 6, x = pix & 63;
                const float* omr = Af + (size_t)m * 216;
                float dy = omr[g * 18 + tap * 2 + 0];
                float dx = omr[g * 18 + tap * 2 + 1];
                float ml = omr[144 + g * 9 + tap];
                float mask = 1.f / (1.f + __expf(-ml));
                float sy = (float)(y + kyo) + dy;
                float sx = (float)(x + kxo) + dx;
                float y0f = floorf(sy), x0f = floorf(sx);
                float wy = sy - y0f, wx = sx - x0f;
                int iy0 = (int)y0f, ix0 = (int)x0f;
                const float* cb = coarseUp + ((size_t)b << 12) * 256;
                float v00 = 0.f, v01 = 0.f, v10 = 0.f, v11 = 0.f;
                bool x0ok = (unsigned)ix0 < 64u, x1ok = (unsigned)(ix0 + 1) < 64u;
                if ((unsigned)iy0 < 64u) {
                    const float* row = cb + (size_t)iy0 * 64 * 256;
                    if (x0ok) v00 = row[(size_t)ix0 * 256 + ch];
                    if (x1ok) v01 = row[(size_t)(ix0 + 1) * 256 + ch];
                }
                if ((unsigned)(iy0 + 1) < 64u) {
                    const float* row = cb + (size_t)(iy0 + 1) * 64 * 256;
                    if (x0ok) v10 = row[(size_t)ix0 * 256 + ch];
                    if (x1ok) v11 = row[(size_t)(ix0 + 1) * 256 + ch];
                }
                float top = v00 * (1.f - wx) + v01 * wx;
                float bot = v10 * (1.f - wx) + v11 * wx;
                As[r][atc] = f2bfbits((top * (1.f - wy) + bot * wy) * mask);
            }
        } else if (A_MODE == 5) {
            // 3x3 im2col over bf16 align [16384][512]; flat k = tap*512 + c
            const int chunk = kb >> 5;
            const int tap = chunk >> 4;
            const int cc  = (chunk & 15) * 32 + atc;
            const int dy = tap / 3 - 1, dx = tap % 3 - 1;
            #pragma unroll 1
            for (int rr = 0; rr < 16; ++rr) {
                int r = atr + rr * 8, m = m0 + r;
                int b = m >> 12, pix = m & 4095;
                int y = (pix >> 6) + dy, x = (pix & 63) + dx;
                unsigned short v = 0;
                if ((unsigned)y < 64u && (unsigned)x < 64u)
                    v = Ab[((size_t)((b << 12) + y * 64 + x)) * 512 + cc];
                As[r][atc] = v;
            }
        } else {
            #pragma unroll 1
            for (int rr = 0; rr < 16; ++rr) {
                int r = atr + rr * 8, m = m0 + r;
                int k = kb + atc;
                if (A_MODE == 0) {
                    As[r][atc] = f2bfbits(Af[(size_t)m * K + k]);
                } else if (A_MODE == 1) {
                    int b = m >> 12;
                    As[r][atc] =
                        f2bfbits(Af[(size_t)m * K + k] * (attn[b * 256 + k] + 1.f));
                } else if (A_MODE == 2) {
                    float v = (k < 256) ? fineCal[(size_t)m * 256 + k]
                                        : 2.f * coarseUp[(size_t)m * 256 + (k - 256)];
                    As[r][atc] = f2bfbits(v);
                } else { // 3: bf16 direct
                    As[r][atc] = Ab[(size_t)m * K + k];
                }
            }
        }

        // B async DMA must land before barrier (each wave waits on its own ops)
        asm volatile("s_wait_asynccnt 0x0" ::: "memory");
        __syncthreads();

        // ---------------- compute: 4x2 WMMA tiles per wave ----------------
        Frag bfr[2];
        #pragma unroll
        for (int ni = 0; ni < 2; ++ni) {
            int col = wn * 32 + ni * 16 + lj;
            bfr[ni].q[0] = *(const uint4*)&Bs[col][lhi * 8];
            bfr[ni].q[1] = *(const uint4*)&Bs[col][16 + lhi * 8];
        }
        #pragma unroll
        for (int mi = 0; mi < 4; ++mi) {
            Frag afr;
            int row = wm * 64 + mi * 16 + lj;
            afr.q[0] = *(const uint4*)&As[row][lhi * 8];
            afr.q[1] = *(const uint4*)&As[row][16 + lhi * 8];
            #pragma unroll
            for (int ni = 0; ni < 2; ++ni)
                acc[mi][ni] = __builtin_amdgcn_wmma_f32_16x16x32_bf16(
                    false, afr.v, false, bfr[ni].v, (short)0, acc[mi][ni],
                    false, false);
        }
    }

    // -------------------- epilogue --------------------
    #pragma unroll
    for (int mi = 0; mi < 4; ++mi) {
        #pragma unroll
        for (int ni = 0; ni < 2; ++ni) {
            int n = n0 + wn * 32 + ni * 16 + lj;
            if (n >= N) continue;
            #pragma unroll
            for (int r = 0; r < 8; ++r) {
                int m = m0 + wm * 64 + mi * 16 + lhi * 8 + r;
                float val = acc[mi][ni][r];
                if (E_MODE == 0) {
                    outF[(size_t)m * N + n] = val;
                } else if (E_MODE == 1) {
                    outB[(size_t)m * N + n] = f2bfbits(val);
                } else if (E_MODE == 2) {
                    outF[(size_t)m * N + n] = val + bias[n];
                } else { // 3: relu(acc+bias) + resid
                    float v2 = fmaxf(val + bias[n], 0.f) + resid[(size_t)m * 256 + n];
                    outF[(size_t)m * N + n] = v2;
                }
            }
        }
    }
}

// ---------------------------------------------------------------------------
extern "C" void kernel_launch(void* const* d_in, const int* in_sizes, int n_in,
                              void* d_out, int out_size, void* d_ws, size_t ws_size,
                              hipStream_t stream)
{
    (void)in_sizes; (void)n_in; (void)out_size; (void)ws_size;

    const float* coarse = (const float*)d_in[0];   // [4,32,32,256]
    const float* fine   = (const float*)d_in[1];   // [4,64,64,256]
    const float* w_att  = (const float*)d_in[2];   // [256,256]
    const float* w_sel  = (const float*)d_in[3];   // [256,256]
    const float* w_off  = (const float*)d_in[4];   // [512,512]
    const float* w_om   = (const float*)d_in[5];   // [4608,216]
    const float* b_om   = (const float*)d_in[6];   // [216]
    const float* w_dcn  = (const float*)d_in[7];   // [2304,256]
    const float* b_dcn  = (const float*)d_in[8];   // [256]
    float* out = (float*)d_out;                    // [16384,256]

    char* ws = (char*)d_ws;
    float* coarse_up = (float*)ws;                  ws += (size_t)16384 * 256 * 4;
    float* fine_cal  = (float*)ws;                  ws += (size_t)16384 * 256 * 4;
    unsigned short* align_bf = (unsigned short*)ws; ws += (size_t)16384 * 512 * 2;
    float* om     = (float*)ws;                     ws += (size_t)16384 * 216 * 4;
    unsigned short* w_selT = (unsigned short*)ws;   ws += (size_t)256 * 256 * 2;
    unsigned short* w_offT = (unsigned short*)ws;   ws += (size_t)512 * 512 * 2;
    unsigned short* w_omT  = (unsigned short*)ws;   ws += (size_t)256 * 4608 * 2;
    unsigned short* w_dcnT = (unsigned short*)ws;   ws += (size_t)256 * 2304 * 2;
    float* pooled = (float*)ws;                     ws += 4 * 256 * 4;
    float* attn   = (float*)ws;                     ws += 4 * 256 * 4;

    // 0) one-time weight transpose + bf16 convert (padded to 128-mult rows)
    wprep_kernel<<<dim3(256, 1),  256, 0, stream>>>(w_sel, w_selT, 256, 256);
    wprep_kernel<<<dim3(512, 2),  256, 0, stream>>>(w_off, w_offT, 512, 512);
    wprep_kernel<<<dim3(256, 18), 256, 0, stream>>>(w_om,  w_omT, 4608, 216);
    wprep_kernel<<<dim3(256, 9),  256, 0, stream>>>(w_dcn, w_dcnT, 2304, 256);

    // 1) global mean pool of fine
    pool_kernel<<<dim3(4, 4), 256, 0, stream>>>(fine, pooled);
    // 2) channel attention (sigmoid of 1x1 conv on pooled)
    attn_kernel<<<4, 256, 0, stream>>>(pooled, w_att, attn);
    // 3) bilinear upsample coarse -> coarse_up
    upsample_kernel<<<dim3(4096, 4), 256, 0, stream>>>(coarse, coarse_up);
    // 4) fine_cal = (fine * (attn+1)) @ w_sel           M=16384 K=256  N=256
    gemm_wmma<1, 0><<<dim3(128, 2), 256, 0, stream>>>(
        fine, nullptr, attn, nullptr, nullptr,
        w_selT, nullptr, nullptr, fine_cal, nullptr, 16384, 256, 256);
    // 5) align(bf16) = concat(fine_cal, 2*coarse_up) @ w_off   K=512 N=512
    gemm_wmma<2, 1><<<dim3(128, 4), 256, 0, stream>>>(
        nullptr, nullptr, nullptr, fine_cal, coarse_up,
        w_offT, nullptr, nullptr, nullptr, align_bf, 16384, 512, 512);
    // 6) om = conv3x3(align) @ w_om + b_om (im2col fused)      K=4608 N=216
    gemm_wmma<5, 2><<<dim3(128, 2), 256, 0, stream>>>(
        nullptr, align_bf, nullptr, nullptr, nullptr,
        w_omT, b_om, nullptr, om, nullptr, 16384, 4608, 216);
    // 7) out = relu(deform_sample(coarse_up; om) @ w_dcn + b_dcn) + fine_cal
    //    (bilinear sampling + sigmoid mask fused into A-tile load)  K=2304 N=256
    gemm_wmma<4, 3><<<dim3(128, 2), 256, 0, stream>>>(
        om, nullptr, nullptr, nullptr, coarse_up,
        w_dcnT, b_dcn, fine_cal, out, nullptr, 16384, 2304, 256);
}
